// GraphAttention_21122649162384
// MI455X (gfx1250) — matrix operand order repaired and checked by reading, not compiled, not run on hardware
//
#include <hip/hip_runtime.h>
#include <hip/hip_bf16.h>
#include <math.h>

// ---------------------------------------------------------------------------
// GraphAttention forward for MI455X (gfx1250, wave32, WMMA bf16).
// Dominant cost: mask = v @ v^T  (3072^3 GEMM, x5 layers) -> v_wmma_f32_16x16x32_bf16.
// ---------------------------------------------------------------------------

typedef __attribute__((ext_vector_type(16))) __bf16 v16bf;
typedef __attribute__((ext_vector_type(8)))  __bf16 v8bf;
typedef __attribute__((ext_vector_type(8)))  float  v8f;

#define GN 3072
#define GF 256
#define GH 64
#define GNH 4
#define GE 8
#define NEG_INF_F (-9.0e15f)

#define MODE_PLAIN 0
#define MODE_MASK  1
#define MODE_HEAD  2
#define MODE_FINAL 3

__device__ __forceinline__ float elu1(float x) {
    return x > 0.0f ? x : (__expf(x) - 1.0f);
}

// ---- fragment loaders (row-major A[M,K], B stored as Bt[N,K] row-major) ----
// A 16x32 bf16 fragment: lane l -> row m0+(l&15); K elems {g*8..g*8+7, 16+g*8..16+g*8+7}, g=l>>4
__device__ __forceinline__ v16bf load_a_frag(const __bf16* __restrict__ A, int lda,
                                             int m0, int k0, int lane) {
    const int r = m0 + (lane & 15);
    const int g = lane >> 4;
    const __bf16* p = A + (size_t)r * lda + k0 + g * 8;
    v8bf lo = *(const v8bf*)p;          // 16B load
    v8bf hi = *(const v8bf*)(p + 16);   // 16B load
    return __builtin_shufflevector(lo, hi, 0,1,2,3,4,5,6,7,8,9,10,11,12,13,14,15);
}

// B 32x16 bf16 fragment: lane l -> col n0+(l&15); K elems {g*8..g*8+7, 16+g*8..16+g*8+7}, g=l>>4
__device__ __forceinline__ v16bf load_b_frag(const __bf16* __restrict__ Bt, int ldb,
                                             int n0, int k0, int lane) {
    const int n = n0 + (lane & 15);
    const int g = lane >> 4;
    const __bf16* p = Bt + (size_t)n * ldb + k0 + g * 8;
    v8bf lo = *(const v8bf*)p;          // 16B load
    v8bf hi = *(const v8bf*)(p + 16);   // 16B load
    return __builtin_shufflevector(lo, hi, 0,1,2,3,4,5,6,7,8,9,10,11,12,13,14,15);
}

// ---------------------------------------------------------------------------
// Generic wave GEMM: C[M,N] = A[M,K] * Bt[N,K]^T, bf16 in / f32 accum.
// One wave (32 threads) per block computes a 32x64 tile: 8 WMMAs per K=32 step.
// Epilogue fused per `mode`.
// ---------------------------------------------------------------------------
__global__ void __launch_bounds__(32)
gat_gemm_abT(const __bf16* __restrict__ A, const __bf16* __restrict__ Bt,
             int M, int N, int K,
             float* __restrict__ out,
             const float* __restrict__ left, const float* __restrict__ right,
             int mode)
{
    const int lane = threadIdx.x;
    const int n0 = blockIdx.x * 64;
    const int m0 = blockIdx.y * 32;

    v8f acc[2][4];
    for (int mi = 0; mi < 2; ++mi)
        for (int ni = 0; ni < 4; ++ni)
            acc[mi][ni] = (v8f){0.f,0.f,0.f,0.f,0.f,0.f,0.f,0.f};

    for (int k0 = 0; k0 < K; k0 += 32) {
        if (k0 + 128 < K) {  // light prefetch into cache hierarchy
            __builtin_prefetch(A  + (size_t)(m0 + (lane & 31)) * K + k0 + 128, 0, 1);
            __builtin_prefetch(Bt + (size_t)(n0 + (lane & 31)) * K + k0 + 128, 0, 1);
        }
        v16bf a0 = load_a_frag(A, K, m0,      k0, lane);
        v16bf a1 = load_a_frag(A, K, m0 + 16, k0, lane);
        v16bf b0 = load_b_frag(Bt, K, n0,      k0, lane);
        v16bf b1 = load_b_frag(Bt, K, n0 + 16, k0, lane);
        v16bf b2 = load_b_frag(Bt, K, n0 + 32, k0, lane);
        v16bf b3 = load_b_frag(Bt, K, n0 + 48, k0, lane);

        acc[0][0] = __builtin_amdgcn_wmma_f32_16x16x32_bf16(false, a0, false, b0, (short)0, acc[0][0], false, false);
        acc[0][1] = __builtin_amdgcn_wmma_f32_16x16x32_bf16(false, a0, false, b1, (short)0, acc[0][1], false, false);
        acc[0][2] = __builtin_amdgcn_wmma_f32_16x16x32_bf16(false, a0, false, b2, (short)0, acc[0][2], false, false);
        acc[0][3] = __builtin_amdgcn_wmma_f32_16x16x32_bf16(false, a0, false, b3, (short)0, acc[0][3], false, false);
        acc[1][0] = __builtin_amdgcn_wmma_f32_16x16x32_bf16(false, a1, false, b0, (short)0, acc[1][0], false, false);
        acc[1][1] = __builtin_amdgcn_wmma_f32_16x16x32_bf16(false, a1, false, b1, (short)0, acc[1][1], false, false);
        acc[1][2] = __builtin_amdgcn_wmma_f32_16x16x32_bf16(false, a1, false, b2, (short)0, acc[1][2], false, false);
        acc[1][3] = __builtin_amdgcn_wmma_f32_16x16x32_bf16(false, a1, false, b3, (short)0, acc[1][3], false, false);
    }

    // C/D layout: VGPR i of lane l holds element (row = i + 8*(l>>4), col = l&15) of a 16x16 tile
    const int colb = lane & 15;
    const int rsub = (lane >> 4) * 8;
    for (int mi = 0; mi < 2; ++mi) {
        for (int ni = 0; ni < 4; ++ni) {
            const int col = n0 + ni * 16 + colb;
            const int rb  = m0 + mi * 16 + rsub;
            v8f c = acc[mi][ni];
            #pragma unroll
            for (int i = 0; i < 8; ++i) {
                const int row = rb + i;
                float val = c[i];
                float o;
                if (mode == MODE_MASK) {
                    float z  = left[row] + right[col];
                    float lr = (z >= 0.0f) ? z : 0.5f * z;   // leaky_relu slope 0.5
                    o = (val > 0.0f) ? lr : NEG_INF_F;
                } else if (mode == MODE_HEAD) {
                    o = elu1(val);
                } else if (mode == MODE_FINAL) {
                    o = elu1(elu1(val));                      // elu applied twice (as in ref)
                } else {
                    o = val;
                }
                out[(size_t)row * N + col] = o;
            }
        }
    }
}

// ---------------------------------------------------------------------------
// v[n,m] = sum_e adj[n,m,e] * w[e]  (bandwidth-bound; 32B/thread coalesced)
// ---------------------------------------------------------------------------
__global__ void gat_adj_project(const float* __restrict__ adj,
                                const float* __restrict__ w,
                                __bf16* __restrict__ v, long long total)
{
    long long idx = (long long)blockIdx.x * blockDim.x + threadIdx.x;
    if (idx >= total) return;
    const float4* a = (const float4*)(adj + idx * GE);
    float4 a0 = a[0], a1 = a[1];
    float r = a0.x * w[0] + a0.y * w[1] + a0.z * w[2] + a0.w * w[3]
            + a1.x * w[4] + a1.y * w[5] + a1.z * w[6] + a1.w * w[7];
    v[idx] = (__bf16)r;
}

// ---------------------------------------------------------------------------
// Row softmax over masked logits S (f32) -> P (bf16)
// ---------------------------------------------------------------------------
__global__ void gat_softmax_rows(const float* __restrict__ S,
                                 __bf16* __restrict__ P, int N)
{
    __shared__ float red[256];
    const int row = blockIdx.x;
    const int tid = threadIdx.x;
    const float* s = S + (size_t)row * N;

    float m = -INFINITY;
    for (int j = tid; j < N; j += 256) m = fmaxf(m, s[j]);
    red[tid] = m; __syncthreads();
    for (int off = 128; off > 0; off >>= 1) {
        if (tid < off) red[tid] = fmaxf(red[tid], red[tid + off]);
        __syncthreads();
    }
    m = red[0]; __syncthreads();

    float sum = 0.0f;
    for (int j = tid; j < N; j += 256) sum += __expf(s[j] - m);
    red[tid] = sum; __syncthreads();
    for (int off = 128; off > 0; off >>= 1) {
        if (tid < off) red[tid] += red[tid + off];
        __syncthreads();
    }
    const float inv = 1.0f / red[0];

    __bf16* p = P + (size_t)row * N;
    for (int j = tid; j < N; j += 256)
        p[j] = (__bf16)(__expf(s[j] - m) * inv);
}

// ---------------------------------------------------------------------------
// left[i] = Wh[i,:] . a[0:64], right[i] = Wh[i,:] . a[64:128]; also WhT bf16
// ---------------------------------------------------------------------------
__global__ void gat_prep(const float* __restrict__ Wh, const float* __restrict__ a,
                         float* __restrict__ left, float* __restrict__ right,
                         __bf16* __restrict__ WhT, int M)
{
    __shared__ float lbuf[GH], rbuf[GH];
    const int row = blockIdx.x;
    const int t = threadIdx.x;           // 64 threads
    float w = Wh[(size_t)row * GH + t];
    WhT[(size_t)t * M + row] = (__bf16)w;
    lbuf[t] = w * a[t];
    rbuf[t] = w * a[GH + t];
    __syncthreads();
    for (int off = 32; off > 0; off >>= 1) {
        if (t < off) { lbuf[t] += lbuf[t + off]; rbuf[t] += rbuf[t + off]; }
        __syncthreads();
    }
    if (t == 0) { left[row] = lbuf[0]; right[row] = rbuf[0]; }
}

// f32 -> bf16 copy
__global__ void gat_cvt_bf16(const float* __restrict__ in, __bf16* __restrict__ out, int n)
{
    int i = blockIdx.x * blockDim.x + threadIdx.x;
    if (i < n) out[i] = (__bf16)in[i];
}

// W f32 [R,C] -> Wt bf16 [C,R]
__global__ void gat_transpose_cvt(const float* __restrict__ W, __bf16* __restrict__ Wt,
                                  int R, int C)
{
    int i = blockIdx.x * blockDim.x + threadIdx.x;
    if (i >= R * C) return;
    int r = i / C, c = i % C;
    Wt[(size_t)c * R + r] = (__bf16)W[i];
}

// x_cat = bf16(elu(concat(heads)))   heads: [4][N][64]
__global__ void gat_build_xcat(const float* __restrict__ heads, __bf16* __restrict__ xcat, int M)
{
    int idx = blockIdx.x * blockDim.x + threadIdx.x;
    if (idx >= M * GF) return;
    int i = idx >> 8;            // node
    int c = idx & 255;           // 0..255
    int h = c >> 6, f = c & 63;
    float x = heads[(size_t)h * M * GH + (size_t)i * GH + f];
    xcat[idx] = (__bf16)elu1(x);
}

// ---------------------------------------------------------------------------
extern "C" void kernel_launch(void* const* d_in, const int* in_sizes, int n_in,
                              void* d_out, int out_size, void* d_ws, size_t ws_size,
                              hipStream_t stream)
{
    (void)in_sizes; (void)n_in; (void)out_size; (void)ws_size;
    const float* x      = (const float*)d_in[0];   // [N, 256]
    const float* adj    = (const float*)d_in[1];   // [N, N, 8]
    const float* W_h    = (const float*)d_in[2];   // [4, 256, 64]
    const float* a_h    = (const float*)d_in[3];   // [4, 128, 1]
    const float* adjw_h = (const float*)d_in[4];   // [4, 8, 1]
    const float* W_f    = (const float*)d_in[5];   // [256, 64]
    const float* a_f    = (const float*)d_in[6];   // [128, 1]
    const float* adjw_f = (const float*)d_in[7];   // [8, 1]
    float* out = (float*)d_out;                    // [N, 64]

    const int N = GN, F = GF, H = GH;

    // ---- workspace carve-out (~83 MB) ----
    char* ws = (char*)d_ws;
    auto carve = [&](size_t bytes) -> char* {
        char* p = ws; ws += (bytes + 255) & ~(size_t)255; return p;
    };
    __bf16* x_bf    = (__bf16*)carve((size_t)N * F * 2);
    __bf16* xcat_bf = (__bf16*)carve((size_t)N * F * 2);
    __bf16* Wt      = (__bf16*)carve((size_t)H * F * 2);
    float*  Wh      = (float*) carve((size_t)N * H * 4);
    __bf16* WhT     = (__bf16*)carve((size_t)H * N * 2);
    float*  left    = (float*) carve((size_t)N * 4);
    float*  right   = (float*) carve((size_t)N * 4);
    __bf16* v       = (__bf16*)carve((size_t)N * N * 2);
    float*  S       = (float*) carve((size_t)N * N * 4);
    __bf16* P       = (__bf16*)carve((size_t)N * N * 2);
    float*  heads   = (float*) carve((size_t)GNH * N * H * 4);

    const long long NN = (long long)N * N;
    const dim3 gWh(H / 64, N / 32);     // [N,64] = [N,K] x [64,K]^T
    const dim3 gMask(N / 64, N / 32);   // [N,N]  = v x v^T

    gat_cvt_bf16<<<(N * F + 255) / 256, 256, 0, stream>>>(x, x_bf, N * F);

    for (int h = 0; h < GNH; ++h) {
        gat_transpose_cvt<<<(F * H + 255) / 256, 256, 0, stream>>>(
            W_h + (size_t)h * F * H, Wt, F, H);
        gat_gemm_abT<<<gWh, 32, 0, stream>>>(x_bf, Wt, N, H, F, Wh,
                                             nullptr, nullptr, MODE_PLAIN);
        gat_prep<<<N, H, 0, stream>>>(Wh, a_h + (size_t)h * 2 * H, left, right, WhT, N);
        gat_adj_project<<<(unsigned)((NN + 255) / 256), 256, 0, stream>>>(
            adj, adjw_h + (size_t)h * GE, v, NN);
        gat_gemm_abT<<<gMask, 32, 0, stream>>>(v, v, N, N, N, S,
                                               left, right, MODE_MASK);
        gat_softmax_rows<<<N, 256, 0, stream>>>(S, P, N);
        gat_gemm_abT<<<gWh, 32, 0, stream>>>(P, WhT, N, H, N,
                                             heads + (size_t)h * N * H,
                                             nullptr, nullptr, MODE_HEAD);
    }

    gat_build_xcat<<<(N * F + 255) / 256, 256, 0, stream>>>(heads, xcat_bf, N);

    // final layer
    gat_transpose_cvt<<<(F * H + 255) / 256, 256, 0, stream>>>(W_f, Wt, F, H);
    gat_gemm_abT<<<gWh, 32, 0, stream>>>(xcat_bf, Wt, N, H, F, Wh,
                                         nullptr, nullptr, MODE_PLAIN);
    gat_prep<<<N, H, 0, stream>>>(Wh, a_f, left, right, WhT, N);
    gat_adj_project<<<(unsigned)((NN + 255) / 256), 256, 0, stream>>>(adj, adjw_f, v, NN);
    gat_gemm_abT<<<gMask, 32, 0, stream>>>(v, v, N, N, N, S, left, right, MODE_MASK);
    gat_softmax_rows<<<N, 256, 0, stream>>>(S, P, N);
    gat_gemm_abT<<<gWh, 32, 0, stream>>>(P, WhT, N, H, N, out,
                                         nullptr, nullptr, MODE_FINAL);
}
